// VarianceAdaptor_37245956391153
// MI455X (gfx1250) — matrix operand order repaired and checked by reading, not compiled, not run on hardware
//
#include <hip/hip_runtime.h>
#include <hip/hip_bf16.h>

typedef __attribute__((ext_vector_type(16))) _Float16 v16h;
typedef __attribute__((ext_vector_type(8)))  _Float16 v8h;
typedef __attribute__((ext_vector_type(8)))  float    v8f;

#define NB_  16      // batch
#define NL_  512     // encoder length
#define NC_  256     // channels E == F
#define NM_  (NB_*NL_)   // 8192 GEMM rows
#define LDEC_ 4096   // decoder length
#define NBINS1_ 255  // N_BINS-1

// ---------------------------------------------------------------- bucketize
__device__ __forceinline__ int bucketize_left(const float* bins, float v) {
    int lo = 0, hi = NBINS1_;
    while (lo < hi) { int mid = (lo + hi) >> 1; if (bins[mid] < v) lo = mid + 1; else hi = mid; }
    return lo;                          // first idx with bins[idx] >= v  (side='left')
}

// ------------------------------------------------------- embeddings + adds
// grid = NM_ blocks, 256 threads; one (b,l) row per block, one channel/thread
__global__ void embed_kernel(const float* __restrict__ enc,
                             const float* __restrict__ pitch,
                             const float* __restrict__ energy,
                             const unsigned char* __restrict__ mask,
                             const float* __restrict__ pbins,
                             const float* __restrict__ ebins,
                             const float* __restrict__ ptab,
                             const float* __restrict__ etab,
                             float* __restrict__ enc1,
                             float* __restrict__ enc2) {
    int bl = blockIdx.x, c = threadIdx.x;
    bool m = mask[bl] != 0;
    int pi = bucketize_left(pbins, pitch[bl]);
    int ei = bucketize_left(ebins, energy[bl]);
    float pe = m ? 0.f : ptab[pi * NC_ + c];
    float ee = m ? 0.f : etab[ei * NC_ + c];
    float e1 = enc[(size_t)bl * NC_ + c] + pe;
    enc1[(size_t)bl * NC_ + c] = e1;
    enc2[(size_t)bl * NC_ + c] = e1 + ee;
}

// ------------------------------------------- pack weights into B-fragments
// Dest layout: P[((nt*8+kb)*32 + lane)*16 + e] = W[kb*32 + 8*half + 16*(e>=8) + (e&7)][nt*16 + (lane&15)]
__global__ void packw_kernel(const float* __restrict__ W, _Float16* __restrict__ P) {
    int idx = blockIdx.x * blockDim.x + threadIdx.x;   // 16*8*32 = 4096 total
    if (idx >= 4096) return;
    int lane = idx & 31, kb = (idx >> 5) & 7, nt = idx >> 8;
    int n = nt * 16 + (lane & 15), half = lane >> 4;
    _Float16* dst = P + ((size_t)((nt * 8 + kb) * 32 + lane)) * 16;
#pragma unroll
    for (int e = 0; e < 16; ++e)
        dst[e] = (_Float16)W[(size_t)(kb * 32 + 8 * half + ((e >> 3) << 4) + (e & 7)) * NC_ + n];
}

// ------------------------------------------- depthwise conv K=3 (zero pad)
// grid = NM_ blocks, 256 threads; output stored f16 as the GEMM A matrix
__global__ void dwconv_kernel(const float* __restrict__ X,
                              const float* __restrict__ dw,   // (3,1,256)
                              const float* __restrict__ db,
                              _Float16* __restrict__ A) {
    int bl = blockIdx.x, c = threadIdx.x;
    int l = bl & (NL_ - 1);
    float w0 = dw[c], w1 = dw[NC_ + c], w2 = dw[2 * NC_ + c];
    float acc = db[c] + X[(size_t)bl * NC_ + c] * w1;
    if (l > 0)        acc += X[(size_t)(bl - 1) * NC_ + c] * w0;
    if (l < NL_ - 1)  acc += X[(size_t)(bl + 1) * NC_ + c] * w2;
    A[(size_t)bl * NC_ + c] = (_Float16)acc;
}

// -------------------------------------------------- WMMA GEMM + bias + ReLU
// 4 waves/block, each wave owns one 16-row M tile and sweeps all 16 N tiles.
// grid = NM_/64 = 128 blocks of 128 threads.
__global__ void __launch_bounds__(128) gemm_relu_kernel(const _Float16* __restrict__ A,
                                                        const _Float16* __restrict__ P,
                                                        const float* __restrict__ bias,
                                                        float* __restrict__ Z) {
    int lane = threadIdx.x & 31;
    int wave = threadIdx.x >> 5;
    int mtile = blockIdx.x * 4 + wave;
    int row = lane & 15, half = lane >> 4;

    // A fragments: element e holds A[m][kb*32 + 8*half + 16*(e>=8) + (e&7)]
    const _Float16* Abase = A + (size_t)(mtile * 16 + row) * NC_ + half * 8;
    v16h a[8];
#pragma unroll
    for (int kb = 0; kb < 8; ++kb) {
        v8h lo = *(const v8h*)(Abase + kb * 32);
        v8h hi = *(const v8h*)(Abase + kb * 32 + 16);
        v16h t;
#pragma unroll
        for (int e = 0; e < 8; ++e) { t[e] = lo[e]; t[e + 8] = hi[e]; }
        a[kb] = t;
    }

    const v16h* Bp = (const v16h*)P;
#pragma unroll
    for (int nt = 0; nt < 16; ++nt) {
        v8f acc = {};
#pragma unroll
        for (int kb = 0; kb < 8; ++kb) {
            v16h bf = Bp[(size_t)((nt * 8 + kb) * 32 + lane)];
            acc = __builtin_amdgcn_wmma_f32_16x16x32_f16(
                false, a[kb], false, bf, (short)0, acc, false, false);
        }
        int n = nt * 16 + row;
        float bv = bias[n];
#pragma unroll
        for (int r = 0; r < 8; ++r) {
            float z = acc[r] + bv;
            Z[(size_t)(mtile * 16 + r + 8 * half) * NC_ + n] = fmaxf(z, 0.0f);
        }
    }
}

// ------------------------------------------------------------- LayerNorm
// grid = NM_ blocks, 256 threads (one row per block)
__global__ void ln_kernel(const float* __restrict__ Z,
                          const float* __restrict__ g,
                          const float* __restrict__ be,
                          float* __restrict__ H) {
    int m = blockIdx.x, c = threadIdx.x;
    float x = Z[(size_t)m * NC_ + c];
    float s = x, s2 = x * x;
#pragma unroll
    for (int off = 16; off > 0; off >>= 1) {
        s  += __shfl_down(s,  off, 32);
        s2 += __shfl_down(s2, off, 32);
    }
    __shared__ float sh1[8], sh2[8];
    int wave = c >> 5, lane = c & 31;
    if (lane == 0) { sh1[wave] = s; sh2[wave] = s2; }
    __syncthreads();
    if (wave == 0) {
        float a1 = (lane < 8) ? sh1[lane] : 0.f;
        float a2 = (lane < 8) ? sh2[lane] : 0.f;
#pragma unroll
        for (int off = 4; off > 0; off >>= 1) {
            a1 += __shfl_down(a1, off, 32);
            a2 += __shfl_down(a2, off, 32);
        }
        if (lane == 0) { sh1[0] = a1; sh2[0] = a2; }
    }
    __syncthreads();
    float mean = sh1[0] * (1.f / NC_);
    float var  = sh2[0] * (1.f / NC_) - mean * mean;
    float inv  = rsqrtf(var + 1e-5f);
    H[(size_t)m * NC_ + c] = (x - mean) * inv * g[c] + be[c];
}

// ----------------------------------------------------- final 256->1 linear
__global__ void linear_kernel(const float* __restrict__ H,
                              const float* __restrict__ lw,
                              const float* __restrict__ lb,
                              const unsigned char* __restrict__ mask,
                              float* __restrict__ out) {
    int m = blockIdx.x, c = threadIdx.x;
    float v = H[(size_t)m * NC_ + c] * lw[c];
#pragma unroll
    for (int off = 16; off > 0; off >>= 1) v += __shfl_down(v, off, 32);
    __shared__ float sh[8];
    int wave = c >> 5, lane = c & 31;
    if (lane == 0) sh[wave] = v;
    __syncthreads();
    if (c == 0) {
        float t = 0.f;
#pragma unroll
        for (int i = 0; i < 8; ++i) t += sh[i];
        out[m] = mask[m] ? 0.f : (t + lb[0]);
    }
}

// -------------------------------------- length regulator: cumsum + scatter
// grid = NB_ blocks of 512 threads
__global__ void lr_map_kernel(const int* __restrict__ dur, int* __restrict__ map) {
    int b = blockIdx.x, l = threadIdx.x;
    for (int i = l; i < LDEC_; i += NL_) map[(size_t)b * LDEC_ + i] = -1;
    __shared__ int cs[NL_];
    int d = dur[b * NL_ + l];
    cs[l] = d;
    __syncthreads();
    for (int off = 1; off < NL_; off <<= 1) {
        int v = (l >= off) ? cs[l - off] : 0;
        __syncthreads();
        cs[l] += v;
        __syncthreads();
    }
    int hi = cs[l], lo = hi - d;
    for (int dd = lo; dd < hi && dd < LDEC_; ++dd)
        if (dd >= 0) map[(size_t)b * LDEC_ + dd] = l;
}

// grid = NB_*LDEC_ blocks of 64 threads (float4 per thread)
__global__ void gather_kernel(const float* __restrict__ enc2,
                              const int* __restrict__ map,
                              float* __restrict__ out) {
    int bd = blockIdx.x;
    int b = bd >> 12;           // / LDEC_
    int l = map[bd];
    float4 v = make_float4(0.f, 0.f, 0.f, 0.f);
    if (l >= 0)
        v = ((const float4*)(enc2 + ((size_t)b * NL_ + l) * NC_))[threadIdx.x];
    ((float4*)(out + (size_t)bd * NC_))[threadIdx.x] = v;
}

// ---------------------------------------------------------------- launcher
struct PredP {
    const float *dw1, *db1, *pw1, *pb1, *g1, *be1;
    const float *dw2, *db2, *pw2, *pb2, *g2, *be2;
    const float *lw, *lb;
};

static PredP load_pred(void* const* d_in, int base) {
    PredP p;
    p.dw1 = (const float*)d_in[base + 0];  p.db1 = (const float*)d_in[base + 1];
    p.pw1 = (const float*)d_in[base + 2];  p.pb1 = (const float*)d_in[base + 3];
    p.g1  = (const float*)d_in[base + 4];  p.be1 = (const float*)d_in[base + 5];
    p.dw2 = (const float*)d_in[base + 6];  p.db2 = (const float*)d_in[base + 7];
    p.pw2 = (const float*)d_in[base + 8];  p.pb2 = (const float*)d_in[base + 9];
    p.g2  = (const float*)d_in[base + 10]; p.be2 = (const float*)d_in[base + 11];
    p.lw  = (const float*)d_in[base + 12]; p.lb  = (const float*)d_in[base + 13];
    return p;
}

static void run_predictor(const float* X, const PredP& p,
                          const _Float16* pw1P, const _Float16* pw2P,
                          _Float16* Aconv, float* Zbuf, float* Hbuf,
                          const unsigned char* mask, float* out, hipStream_t stream) {
    dwconv_kernel<<<NM_, NC_, 0, stream>>>(X, p.dw1, p.db1, Aconv);
    gemm_relu_kernel<<<NM_ / 64, 128, 0, stream>>>(Aconv, pw1P, p.pb1, Zbuf);
    ln_kernel<<<NM_, NC_, 0, stream>>>(Zbuf, p.g1, p.be1, Hbuf);
    dwconv_kernel<<<NM_, NC_, 0, stream>>>(Hbuf, p.dw2, p.db2, Aconv);
    gemm_relu_kernel<<<NM_ / 64, 128, 0, stream>>>(Aconv, pw2P, p.pb2, Zbuf);
    ln_kernel<<<NM_, NC_, 0, stream>>>(Zbuf, p.g2, p.be2, Hbuf);
    linear_kernel<<<NM_, NC_, 0, stream>>>(Hbuf, p.lw, p.lb, mask, out);
}

extern "C" void kernel_launch(void* const* d_in, const int* in_sizes, int n_in,
                              void* d_out, int out_size, void* d_ws, size_t ws_size,
                              hipStream_t stream) {
    const float*         enc_in   = (const float*)d_in[0];
    const int*           dur      = (const int*)d_in[1];
    const float*         pitch    = (const float*)d_in[2];
    const float*         energy   = (const float*)d_in[3];
    const unsigned char* mask     = (const unsigned char*)d_in[4];
    // d_in[5] = decoder_seq_len (device scalar) -> 4096 in this configuration
    const float* pbins = (const float*)d_in[6];
    const float* ebins = (const float*)d_in[7];
    const float* ptab  = (const float*)d_in[8];
    const float* etab  = (const float*)d_in[9];
    PredP pp = load_pred(d_in, 10);
    PredP ep = load_pred(d_in, 24);
    PredP dp = load_pred(d_in, 38);

    // ---- workspace carve-up (all offsets 256B aligned)
    char* ws = (char*)d_ws;
    float* enc1 = (float*)ws;                         ws += (size_t)NM_ * NC_ * 4;   // 8 MB
    float* enc2 = (float*)ws;                         ws += (size_t)NM_ * NC_ * 4;   // 8 MB
    float* Zbuf = (float*)ws;                         ws += (size_t)NM_ * NC_ * 4;   // 8 MB
    float* Hbuf = (float*)ws;                         ws += (size_t)NM_ * NC_ * 4;   // 8 MB
    _Float16* Aconv = (_Float16*)ws;                  ws += (size_t)NM_ * NC_ * 2;   // 4 MB
    _Float16* packW = (_Float16*)ws;                  ws += (size_t)6 * NC_ * NC_ * 2; // 768 KB
    int* map = (int*)ws;                              ws += (size_t)NB_ * LDEC_ * 4; // 256 KB

    // ---- output carve-up: (encodings, dur, pitch, energy)
    float* out_enc    = (float*)d_out;
    float* out_dur    = out_enc + (size_t)NB_ * LDEC_ * NC_;
    float* out_pitch  = out_dur + NM_;
    float* out_energy = out_pitch + NM_;

    // 1) pre-swizzle the six pointwise weight matrices into WMMA B layout
    const float* Ws[6] = { pp.pw1, pp.pw2, ep.pw1, ep.pw2, dp.pw1, dp.pw2 };
    for (int i = 0; i < 6; ++i)
        packw_kernel<<<16, 256, 0, stream>>>(Ws[i], packW + (size_t)i * NC_ * NC_);

    // 2) variance embeddings
    embed_kernel<<<NM_, NC_, 0, stream>>>(enc_in, pitch, energy, mask,
                                          pbins, ebins, ptab, etab, enc1, enc2);

    // 3) three predictors
    run_predictor(enc1, pp, packW + 0 * (size_t)NC_ * NC_, packW + 1 * (size_t)NC_ * NC_,
                  Aconv, Zbuf, Hbuf, mask, out_pitch, stream);
    run_predictor(enc2, ep, packW + 2 * (size_t)NC_ * NC_, packW + 3 * (size_t)NC_ * NC_,
                  Aconv, Zbuf, Hbuf, mask, out_energy, stream);
    run_predictor(enc2, dp, packW + 4 * (size_t)NC_ * NC_, packW + 5 * (size_t)NC_ * NC_,
                  Aconv, Zbuf, Hbuf, mask, out_dur, stream);

    // 4) length regulation as gather
    lr_map_kernel<<<NB_, NL_, 0, stream>>>(dur, map);
    gather_kernel<<<NB_ * LDEC_, 64, 0, stream>>>(enc2, map, out_enc);
}
